// RNN_6184752906904
// MI455X (gfx1250) — matrix operand order repaired and checked by reading, not compiled, not run on hardware
//
#include <hip/hip_runtime.h>

// ---------------- types for WMMA ----------------
typedef __bf16 bf16_t;
typedef bf16_t v16bf __attribute__((ext_vector_type(16)));
typedef float  v8f   __attribute__((ext_vector_type(8)));

union FragU {
    v16bf v;
    uint4 q[2];
    unsigned u[8];
};

#if __has_builtin(__builtin_amdgcn_global_load_async_to_lds_b128)
#define HAVE_ASYNC_LDS 1
typedef int v4i __attribute__((ext_vector_type(4)));
typedef __attribute__((address_space(1))) v4i gv4i;
typedef __attribute__((address_space(3))) v4i lv4i;
#else
#define HAVE_ASYNC_LDS 0
#endif

// round-to-nearest-even float -> bf16 (bit-level)
__device__ __forceinline__ unsigned short f2bf(float f) {
    union { float f; unsigned u; } c; c.f = f;
    unsigned u = c.u;
    unsigned lsb = (u >> 16) & 1u;
    u += 0x7fffu + lsb;
    return (unsigned short)(u >> 16);
}
__device__ __forceinline__ unsigned pack2bf(float a, float b) {
    return (unsigned)f2bf(a) | ((unsigned)f2bf(b) << 16);
}

// A-fragment K offset for 16-bit A (16x32): VGPR v, lane-half h
__device__ __forceinline__ int kA(int v, int half) {
    return ((v & 3) << 1) + ((v >> 2) << 4) + (half << 3);
}

// ---------------- weight packing: f32 [G][Ksrc] -> bf16 WMMA-B fragments ----------------
__global__ void pack_weights_k(const float* __restrict__ src, unsigned* __restrict__ dst,
                               int Ksrc, int nks, int total) {
    int id = blockIdx.x * 256 + threadIdx.x;
    if (id >= total) return;
    int v    = id & 7;
    int lane = (id >> 3) & 31;
    int rem  = id >> 8;                 // = nt*nks + ks
    int ks   = rem % nks;
    int nt   = rem / nks;
    int n    = lane & 15;
    int half = lane >> 4;
    int g    = nt * 16 + n;
    int k    = ks * 32 + kA(v, half);
    float x0 = (k     < Ksrc) ? src[g * Ksrc + k]     : 0.f;
    float x1 = (k + 1 < Ksrc) ? src[g * Ksrc + k + 1] : 0.f;
    dst[id] = pack2bf(x0, x1);
}

// ---------------- activation packing: build A-fragments once per (t, ks) ----------------
// layer==1: from x [8][1000][240], time-reversed rows for bb>=8, zero-pad K to 256
// layer==2: from hs [1000][16][512] via bidirectional concat formula, K=1024
__global__ void pack_A_k(const float* __restrict__ src, unsigned* __restrict__ dst,
                         int layer, int nks, int total) {
    int id = blockIdx.x * 256 + threadIdx.x;
    if (id >= total) return;
    int v    = id & 7;
    int lane = (id >> 3) & 31;
    int rem  = id >> 8;                 // = t*nks + ks
    int ks   = rem % nks;
    int t    = rem / nks;
    int half = lane >> 4;
    int bb   = lane & 15;               // A row
    int k    = ks * 32 + kA(v, half);
    float x0, x1;
    if (layer == 1) {
        int teff = (bb < 8) ? t : (999 - t);
        int b8 = bb & 7;
        if (k < 240) {
            const float* p = src + (b8 * 1000 + teff) * 240 + k;
            x0 = p[0]; x1 = p[1];
        } else { x0 = 0.f; x1 = 0.f; }
    } else {
        int g = k, tt, bbb, gg;
        if (g < 512) { gg = g;       if (bb < 8) { tt = t;       bbb = bb;     }
                       else          { tt = 999 - t; bbb = bb - 8; } }
        else         { gg = g - 512; if (bb < 8) { tt = 999 - t; bbb = bb + 8; }
                       else          { tt = t;       bbb = bb;     } }
        const float* p = src + (tt * 16 + bbb) * 512 + gg;
        x0 = p[0]; x1 = p[1];
    }
    dst[id] = pack2bf(x0, x1);
}

// ---------------- feed GEMM: pure WMMA on pre-packed fragments ----------------
// Wave handles two N-tiles (gt0, gt1), reusing the A fragment. grid (1000, 4), 256 thr.
__global__ void feed_gemm_k(const unsigned* __restrict__ Ap, const unsigned* __restrict__ Bp,
                            float* __restrict__ Wpre, int nks) {
    int t    = blockIdx.x;
    int wave = threadIdx.x >> 5;
    int lane = threadIdx.x & 31;
    int pair = blockIdx.y * 8 + wave;     // 0..31
    int gt0  = pair * 2, gt1 = pair * 2 + 1;
    int half = lane >> 4;
    int nlo  = lane & 15;
    v8f c0 = {0.f,0.f,0.f,0.f,0.f,0.f,0.f,0.f};
    v8f c1 = {0.f,0.f,0.f,0.f,0.f,0.f,0.f,0.f};

    const uint4* ap  = (const uint4*)(Ap + ((size_t)(t   * nks) * 32 + lane) * 8);
    const uint4* bp0 = (const uint4*)(Bp + ((size_t)(gt0 * nks) * 32 + lane) * 8);
    const uint4* bp1 = (const uint4*)(Bp + ((size_t)(gt1 * nks) * 32 + lane) * 8);

    for (int ks = 0; ks < nks; ++ks) {
        FragU a, b0, b1;
        a.q[0]  = ap[ks * 64];      a.q[1]  = ap[ks * 64 + 1];
        b0.q[0] = bp0[ks * 64];     b0.q[1] = bp0[ks * 64 + 1];
        b1.q[0] = bp1[ks * 64];     b1.q[1] = bp1[ks * 64 + 1];
        c0 = __builtin_amdgcn_wmma_f32_16x16x32_bf16(false, a.v, false, b0.v,
                                                     (short)0, c0, false, false);
        c1 = __builtin_amdgcn_wmma_f32_16x16x32_bf16(false, a.v, false, b1.v,
                                                     (short)0, c1, false, false);
    }
#pragma unroll
    for (int r = 0; r < 8; ++r) {
        int m = r + half * 8;
        Wpre[(size_t)(t * 16 + m) * 1024 + gt0 * 16 + nlo] = c0[r];
        Wpre[(size_t)(t * 16 + m) * 1024 + gt1 * 16 + nlo] = c1[r];
    }
}

// ---------------- deterministic BN stats ----------------
__global__ void stats_partial_k(const float* __restrict__ Wpre,
                                float* __restrict__ psum, float* __restrict__ psq) {
    int blk = blockIdx.x;        // 0..249, each 64 rows of [16000][1024]
    int tid = threadIdx.x;       // 256
    float s[4] = {0,0,0,0}, q[4] = {0,0,0,0};
    const float* base = Wpre + (size_t)blk * 64 * 1024;
    for (int r = 0; r < 64; ++r) {
#pragma unroll
        for (int j = 0; j < 4; ++j) {
            float x = base[r * 1024 + tid + 256 * j];
            s[j] += x; q[j] += x * x;
        }
    }
#pragma unroll
    for (int j = 0; j < 4; ++j) {
        psum[blk * 1024 + tid + 256 * j] = s[j];
        psq [blk * 1024 + tid + 256 * j] = q[j];
    }
}

__global__ void stats_final_k(const float* __restrict__ psum, const float* __restrict__ psq,
                              const float* __restrict__ gamma, const float* __restrict__ beta,
                              float* __restrict__ scale, float* __restrict__ shift) {
    int g = threadIdx.x;         // 1024
    float s = 0.f, q = 0.f;
    for (int b = 0; b < 250; ++b) { s += psum[b * 1024 + g]; q += psq[b * 1024 + g]; }
    float mean = s * (1.0f / 16000.0f);
    float var  = q * (1.0f / 16000.0f) - mean * mean;
    float sc   = gamma[g] * rsqrtf(var + 1e-5f);
    scale[g] = sc;
    shift[g] = beta[g] - mean * sc;
}

// ---------------- recurrence: 1000 sequential steps, one persistent workgroup ----------------
// 32 waves; wave w computes gate columns [32w, 32w+32) as two 16x16 WMMA tiles, K=512.
// Step t's Wpre row (64 KB) is async-copied into the `gates` LDS buffer while WMMAs run.
__global__ void __launch_bounds__(1024, 1)
recurrence_k(const float* __restrict__ Wpre, const float* __restrict__ scale,
             const float* __restrict__ shift, const unsigned* __restrict__ Up,
             float* __restrict__ hs) {
    __shared__ __align__(16) float    h_f32[16 * 512];     // 32 KB
    __shared__ __align__(16) float    gates[16 * 1024];    // 64 KB (also async-copy target)
    __shared__ __align__(16) unsigned hfrag[16 * 32 * 8];  // 16 KB  [ks][lane][v]

    int tid  = threadIdx.x;
    int wave = tid >> 5, lane = tid & 31;
    int half = lane >> 4, nlo = lane & 15;

    for (int i = tid; i < 16 * 512;    i += 1024) h_f32[i] = 0.f;
    for (int i = tid; i < 16 * 32 * 8; i += 1024) hfrag[i] = 0u;
    __syncthreads();

    int nt0 = wave * 2, nt1 = wave * 2 + 1;
    int g0 = nt0 * 16 + nlo, g1 = nt1 * 16 + nlo;
    float sc0 = scale[g0], sh0 = shift[g0];
    float sc1 = scale[g1], sh1 = shift[g1];

    const uint4* ub0 = (const uint4*)(Up + ((size_t)(nt0 * 16) * 32 + lane) * 8);
    const uint4* ub1 = (const uint4*)(Up + ((size_t)(nt1 * 16) * 32 + lane) * 8);

    for (int t = 0; t < 1000; ++t) {
        const float* wrow = Wpre + (size_t)t * 16 * 1024;

#if HAVE_ASYNC_LDS
        // async-copy this step's 64 KB pre-activation row into `gates` (dead during WMMA)
#pragma unroll
        for (int r = 0; r < 4; ++r) {
            int o = wave * 512 + r * 128 + lane * 4;
            __builtin_amdgcn_global_load_async_to_lds_b128(
                (gv4i*)(wrow + o), (lv4i*)(gates + o), 0, 0);
        }
#endif
        if (t < 999) __builtin_prefetch(wrow + 16 * 1024, 0, 0);

        v8f c0 = {0.f,0.f,0.f,0.f,0.f,0.f,0.f,0.f};
        v8f c1 = {0.f,0.f,0.f,0.f,0.f,0.f,0.f,0.f};
#pragma unroll 4
        for (int ks = 0; ks < 16; ++ks) {
            FragU a, b0, b1;
            const uint4* ap = (const uint4*)(hfrag + (ks * 32 + lane) * 8);
            a.q[0]  = ap[0];           a.q[1]  = ap[1];
            b0.q[0] = ub0[ks * 64];    b0.q[1] = ub0[ks * 64 + 1];
            b1.q[0] = ub1[ks * 64];    b1.q[1] = ub1[ks * 64 + 1];
            c0 = __builtin_amdgcn_wmma_f32_16x16x32_bf16(false, a.v, false, b0.v,
                                                         (short)0, c0, false, false);
            c1 = __builtin_amdgcn_wmma_f32_16x16x32_bf16(false, a.v, false, b1.v,
                                                         (short)0, c1, false, false);
        }

#if HAVE_ASYNC_LDS
#if __has_builtin(__builtin_amdgcn_s_wait_asynccnt)
        __builtin_amdgcn_s_wait_asynccnt(0);
#else
        asm volatile("s_wait_asynccnt 0x0" ::: "memory");
#endif
        __syncthreads();   // whole 64 KB row now staged in `gates`
#pragma unroll
        for (int r = 0; r < 8; ++r) {
            int m = r + half * 8;
            gates[m * 1024 + g0] = gates[m * 1024 + g0] * sc0 + sh0 + c0[r];
            gates[m * 1024 + g1] = gates[m * 1024 + g1] * sc1 + sh1 + c1[r];
        }
#else
#pragma unroll
        for (int r = 0; r < 8; ++r) {
            int m = r + half * 8;
            gates[m * 1024 + g0] = wrow[m * 1024 + g0] * sc0 + sh0 + c0[r];
            gates[m * 1024 + g1] = wrow[m * 1024 + g1] * sc1 + sh1 + c1[r];
        }
#endif
        __syncthreads();

        if (wave < 16) {                      // wave w updates h columns [32w, 32w+32)
            float* hsrow = hs + (size_t)t * 8192;
#pragma unroll
            for (int e = 0; e < 16; ++e) {
                int elem = lane + 32 * e;     // 0..511
                int m  = elem >> 5;
                int jj = elem & 31;
                int j  = wave * 32 + jj;
                float aa = gates[m * 1024 + j];
                float zz = gates[m * 1024 + 512 + j];
                float zs = 1.0f / (1.0f + __expf(-zz));
                float hn = zs * h_f32[m * 512 + j] + (1.0f - zs) * fmaxf(aa, 0.0f);
                h_f32[m * 512 + j] = hn;
                hsrow[m * 512 + j] = hn;
                // scatter into A-fragment layout: ks = wave
                int hhalf = (jj >> 3) & 1;
                int v     = ((jj & 7) >> 1) + ((jj >> 4) << 2);
                int lohi  = jj & 1;
                int lanep = hhalf * 16 + m;
                ((unsigned short*)hfrag)[((wave * 32 + lanep) * 8 + v) * 2 + lohi] = f2bf(hn);
            }
        }
        __syncthreads();
    }
}

// ---------------- final output assembly ----------------
__global__ void emit_out_k(const float* __restrict__ hs1, const float* __restrict__ hs2,
                           float* __restrict__ out) {
    int idx = blockIdx.x * 256 + threadIdx.x;
    const int NMAIN = 8192000;
    if (idx < NMAIN) {
        int b   = idx / 1024000;
        int rem = idx % 1024000;
        int t   = rem >> 10;
        int g   = rem & 1023;
        float v = (g < 512) ? hs2[(size_t)(t * 16 + b) * 512 + g]
                            : hs2[(size_t)((999 - t) * 16 + b + 8) * 512 + (g - 512)];
        out[idx] = v;
    } else if (idx < NMAIN + 16384) {
        int i     = idx - NMAIN;
        int b     = i >> 11;
        int which = (i >> 10) & 1;
        int g     = i & 1023;
        const float* hx = which ? hs2 : hs1;
        float v = (g < 512) ? hx[(size_t)(999 * 16 + b) * 512 + g]
                            : hx[(size_t)(b + 8) * 512 + (g - 512)];
        out[idx] = v;
    }
}

// ---------------- host launcher ----------------
extern "C" void kernel_launch(void* const* d_in, const int* in_sizes, int n_in,
                              void* d_out, int out_size, void* d_ws, size_t ws_size,
                              hipStream_t stream) {
    const float* x  = (const float*)d_in[0];
    const float* w0 = (const float*)d_in[1];
    const float* u0 = (const float*)d_in[2];
    const float* g0 = (const float*)d_in[3];
    const float* b0 = (const float*)d_in[4];
    const float* w1 = (const float*)d_in[5];
    const float* u1 = (const float*)d_in[6];
    const float* g1 = (const float*)d_in[7];
    const float* b1 = (const float*)d_in[8];
    float* out = (float*)d_out;

    char* ws = (char*)d_ws;
    size_t off = 0;
    auto take = [&](size_t bytes) -> void* {
        void* p = ws + off;
        off = (off + bytes + 255) & ~(size_t)255;
        return p;
    };
    float*    Wpre = (float*)take((size_t)16384000 * 4);
    float*    hs1  = (float*)take((size_t)8192000 * 4);
    float*    hs2  = (float*)take((size_t)8192000 * 4);
    unsigned* Ap   = (unsigned*)take((size_t)8192000 * 4);   // packed A frags (max: layer 2)
    float*    psum = (float*)take((size_t)250 * 1024 * 4);
    float*    psq  = (float*)take((size_t)250 * 1024 * 4);
    float*    scl  = (float*)take(1024 * 4);
    float*    shf  = (float*)take(1024 * 4);
    unsigned* w0p  = (unsigned*)take((size_t)64 * 8  * 256 * 4);
    unsigned* u0p  = (unsigned*)take((size_t)64 * 16 * 256 * 4);
    unsigned* w1p  = (unsigned*)take((size_t)64 * 32 * 256 * 4);
    unsigned* u1p  = (unsigned*)take((size_t)64 * 16 * 256 * 4);

    // pack weights to bf16 WMMA-B fragment layout
    pack_weights_k<<<(64*8*256 + 255)/256,  256, 0, stream>>>(w0, w0p, 240, 8,  64*8*256);
    pack_weights_k<<<(64*16*256 + 255)/256, 256, 0, stream>>>(u0, u0p, 512, 16, 64*16*256);
    pack_weights_k<<<(64*32*256 + 255)/256, 256, 0, stream>>>(w1, w1p, 1024, 32, 64*32*256);
    pack_weights_k<<<(64*16*256 + 255)/256, 256, 0, stream>>>(u1, u1p, 512, 16, 64*16*256);

    dim3 fg(1000, 4);
    const int totA1 = 1000 * 8 * 256;     // layer-1 A fragments
    const int totA2 = 1000 * 32 * 256;    // layer-2 A fragments

    // ---- layer 1 ----
    pack_A_k<<<(totA1 + 255)/256, 256, 0, stream>>>(x, Ap, 1, 8, totA1);
    feed_gemm_k<<<fg, 256, 0, stream>>>(Ap, w0p, Wpre, 8);
    stats_partial_k<<<250, 256, 0, stream>>>(Wpre, psum, psq);
    stats_final_k<<<1, 1024, 0, stream>>>(psum, psq, g0, b0, scl, shf);
    recurrence_k<<<1, 1024, 0, stream>>>(Wpre, scl, shf, u0p, hs1);

    // ---- layer 2 ----
    pack_A_k<<<(totA2 + 255)/256, 256, 0, stream>>>(hs1, Ap, 2, 32, totA2);
    feed_gemm_k<<<fg, 256, 0, stream>>>(Ap, w1p, Wpre, 32);
    stats_partial_k<<<250, 256, 0, stream>>>(Wpre, psum, psq);
    stats_final_k<<<1, 1024, 0, stream>>>(psum, psq, g1, b1, scl, shf);
    recurrence_k<<<1, 1024, 0, stream>>>(Wpre, scl, shf, u1p, hs2);

    // ---- outputs ----
    emit_out_k<<<(8192000 + 16384) / 256, 256, 0, stream>>>(hs1, hs2, out);
}
